// DynamicGraphConv_BiMamba_54185307406479
// MI455X (gfx1250) — matrix operand order repaired and checked by reading, not compiled, not run on hardware
//
#include <hip/hip_runtime.h>

// ---------------------------------------------------------------------------
// DynamicGraphConv (2 layers) for MI455X / gfx1250.
// B=8, N=512, T=24, D=H=64.  ~30 GFLOP total vs ~400MB traffic @23.3TB/s
// => memory bound; all GEMMs run as v_wmma_f32_16x16x32_bf16 (f32 accum),
// attention fully fused per (b,t) with K/V^T resident in LDS (filled via
// GLOBAL_LOAD_ASYNC_TO_LDS_B128 / ASYNCcnt), softmax row-sums computed by an
// extra WMMA against an all-ones B matrix.
// ---------------------------------------------------------------------------

#define B_ 8
#define N_ 512
#define T_ 24
#define H_ 64
#define SCALE_ 0.125f   // 1/sqrt(64)

typedef __attribute__((ext_vector_type(16))) __bf16 v16bf;
typedef __attribute__((ext_vector_type(8)))  float  v8f;

union V16U { v16bf v; uint4 u[2]; };

__device__ __forceinline__ unsigned short f2bf(float f) {
  // round-to-nearest-even f32 -> bf16
  unsigned int u = __float_as_uint(f);
  unsigned int r = ((u >> 16) & 1u) + 0x7FFFu;
  return (unsigned short)((u + r) >> 16);
}

__device__ __forceinline__ v8f wmma_bf16(v16bf a, v16bf b, v8f c) {
  return __builtin_amdgcn_wmma_f32_16x16x32_bf16(false, a, false, b,
                                                 (short)0, c, false, false);
}

// Async 16B global->LDS copy, tracked by ASYNCcnt (CDNA5 ISA).
// lds_off is the wave-relative LDS byte address (VDST operand).
__device__ __forceinline__ void async_copy_b128(unsigned lds_off, const void* gptr) {
  asm volatile("global_load_async_to_lds_b128 %0, %1, off"
               :: "v"(lds_off), "v"(gptr) : "memory");
}

__device__ __forceinline__ void wait_async0() {
  asm volatile("s_wait_asynccnt 0" ::: "memory");
}

// A-operand 16x32 bf16 (ISA 7.12.2): lane&15 = M row; per-lane halves are
// K chunks {c..c+7, c+16..c+23} with c = (lane>=16 ? 8 : 0). Source row-major.
__device__ __forceinline__ v16bf ldA(const unsigned short* mat, int rowBase,
                                     int colBase, int stride, int lane) {
  int r = rowBase + (lane & 15);
  int c = colBase + ((lane >> 4) << 3);
  const unsigned short* p = mat + r * stride + c;
  V16U o;
  o.u[0] = *(const uint4*)(p);
  o.u[1] = *(const uint4*)(p + 16);
  return o.v;
}

// B-operand 32x16 bf16: lane&15 = N col; K chunks {c..c+7, c+16..c+23}, c = (lane>=16 ? 8:0).
// Source is row-major in the K dimension (i.e. B^T rows contiguous).
__device__ __forceinline__ v16bf ldB(const unsigned short* mat, int rowBase,
                                     int colBase, int stride, int lane) {
  int r = rowBase + (lane & 15);
  int c = colBase + ((lane >> 4) << 3);
  const unsigned short* p = mat + r * stride + c;
  V16U o;
  o.u[0] = *(const uint4*)(p);
  o.u[1] = *(const uint4*)(p + 16);
  return o.v;
}

// A-operand built from fp32 global rows with on-the-fly bf16 conversion.
__device__ __forceinline__ v16bf ldA_f32(const float* row, int colBase, int lane) {
  int c = colBase + ((lane >> 4) << 3);
  unsigned short tmp[16];
#pragma unroll
  for (int i = 0; i < 8; ++i) tmp[i] = f2bf(row[c + i]);
#pragma unroll
  for (int i = 0; i < 8; ++i) tmp[8 + i] = f2bf(row[c + 16 + i]);
  V16U o;
  o.u[0] = *(const uint4*)(&tmp[0]);
  o.u[1] = *(const uint4*)(&tmp[8]);
  return o.v;
}

// ---------------------------------------------------------------------------
// Projection: out = in[row,:]*W + b for W in {Wq,Wk,Wv}.  Rows indexed by
// (b,n,t) through caller-supplied strides so the same kernel serves layer 1
// ([B,N,T,D] input) and layer 2 ([B,T,N,H] input).  Q,K stored bf16
// [bt][n][h]; V stored transposed [bt][h][n] for contiguous attention B-tiles.
// grid = B*T*N/16/4 blocks of 128 threads (4 waves, 16 rows each).
// ---------------------------------------------------------------------------
__global__ __launch_bounds__(128)
void proj_kernel(const float* __restrict__ in,
                 const float* __restrict__ Wq, const float* __restrict__ bq,
                 const float* __restrict__ Wk, const float* __restrict__ bk,
                 const float* __restrict__ Wv, const float* __restrict__ bv,
                 unsigned short* __restrict__ Qo, unsigned short* __restrict__ Ko,
                 unsigned short* __restrict__ Vto,
                 int sB, int sN, int sT) {
  __shared__ unsigned short Wt[3][64 * 64];   // W^T as bf16: Wt[k][h]
  int tid = threadIdx.x;
  {
    const float* Ws[3] = {Wq, Wk, Wv};
#pragma unroll
    for (int m = 0; m < 3; ++m) {
      const float* W = Ws[m];
      for (int i = 0; i < 32; ++i) {
        int idx = tid + i * 128;
        int h = idx >> 6, k = idx & 63;
        Wt[m][k * 64 + h] = f2bf(W[idx]);   // coalesced read, transposed write
      }
    }
  }
  __syncthreads();

  int lane = tid & 31;
  int wave = tid >> 5;
  int w = blockIdx.x * 4 + wave;   // global 16-row tile id, 0..6143
  int bt = w >> 5;                 // 32 tiles per (b,t)
  int tile = w & 31;
  int b = bt / T_, t = bt - b * T_;
  int nBase = tile * 16;

  // A operands: this wave's 16 input rows (f32 -> bf16)
  int n = nBase + (lane & 15);
  const float* rowp = in + (size_t)b * sB + (size_t)n * sN + (size_t)t * sT;
  v16bf a0 = ldA_f32(rowp, 0, lane);
  v16bf a1 = ldA_f32(rowp, 32, lane);

  const float* biases[3] = {bq, bk, bv};
  size_t obase = (size_t)bt * (N_ * H_);

#pragma unroll
  for (int m = 0; m < 3; ++m) {
    const unsigned short* wtm = Wt[m];
#pragma unroll
    for (int kt = 0; kt < 4; ++kt) {
      v8f acc = {};
      v16bf b0 = ldB(wtm, kt * 16, 0, 64, lane);
      v16bf b1 = ldB(wtm, kt * 16, 32, 64, lane);
      acc = wmma_bf16(a0, b0, acc);
      acc = wmma_bf16(a1, b1, acc);
      int col = kt * 16 + (lane & 15);
      float bias = biases[m][col];
      if (m < 2) {
        unsigned short* dst = (m == 0 ? Qo : Ko) + obase;
#pragma unroll
        for (int r = 0; r < 8; ++r) {
          int row = nBase + r + ((lane >> 4) << 3);
          dst[row * H_ + col] = f2bf(acc[r] + bias);
        }
      } else {
        unsigned short* dst = Vto + obase;              // [bt][h][n]
#pragma unroll
        for (int r = 0; r < 8; ++r) {
          int row = nBase + r + ((lane >> 4) << 3);
          dst[col * N_ + row] = f2bf(acc[r] + bias);
        }
      }
    }
  }
}

// ---------------------------------------------------------------------------
// Fused attention for one (b,t): softmax(relu(QK^T*scale)) @ V.
// grid = B*T*4 blocks of 256 threads; block handles 128 i-rows (8 waves x 16).
// LDS: K [512][64] bf16 (64KB) + V^T [64][512] bf16 (64KB) + 1KB/wave P stage,
// filled with GLOBAL_LOAD_ASYNC_TO_LDS_B128 (no VGPR round-trip).
// Row-sum trick: lsum-tile = WMMA(P, ones) replicates sum_j P[i][j] across
// lanes in exactly the C/D layout of the O accumulators -> free denominator.
// ---------------------------------------------------------------------------
__global__ __launch_bounds__(256)
void attn_kernel(const unsigned short* __restrict__ Q,
                 const unsigned short* __restrict__ K,
                 const unsigned short* __restrict__ Vt,
                 float* __restrict__ out,
                 int soB, int soT, int soN, int do_relu) {
  extern __shared__ __align__(16) char smem[];
  unsigned short* Ks  = (unsigned short*)smem;    // 512*64
  unsigned short* Vts = Ks + 512 * 64;            // 64*512
  unsigned short* Ps  = Vts + 64 * 512;           // 8 * 16*32 per-wave scratch

  int tid = threadIdx.x, lane = tid & 31, wave = tid >> 5;
  int bt = blockIdx.x >> 2, quarter = blockIdx.x & 3;
  int b = bt / T_, t = bt - b * T_;
  size_t base = (size_t)bt * (N_ * H_);

  // Async LDS fill: 2 x 64KB of 16B transfers straight into LDS (ASYNCcnt).
  {
    const uint4* gk = (const uint4*)(K + base);
    const uint4* gv = (const uint4*)(Vt + base);
    unsigned ldsBase = (unsigned)(uintptr_t)smem;      // wave-relative LDS byte addr
    for (int i = tid; i < (512 * 64) / 8; i += 256) {
      async_copy_b128(ldsBase + (unsigned)i * 16u, gk + i);
      async_copy_b128(ldsBase + 65536u + (unsigned)i * 16u, gv + i);
    }
    wait_async0();
  }
  __syncthreads();

  int iBase = quarter * 128 + wave * 16;
  v16bf a0 = ldA(Q + base, iBase, 0, 64, lane);
  v16bf a1 = ldA(Q + base, iBase, 32, 64, lane);

  V16U onesu;
  onesu.u[0] = make_uint4(0x3F803F80u, 0x3F803F80u, 0x3F803F80u, 0x3F803F80u);
  onesu.u[1] = onesu.u[0];
  v16bf ones = onesu.v;   // bf16 1.0 x16

  v8f o[4];
#pragma unroll
  for (int ht = 0; ht < 4; ++ht) o[ht] = (v8f){};
  v8f lacc = {};
  unsigned short* pw = Ps + wave * (16 * 32);

  for (int jc = 0; jc < 16; ++jc) {        // 32 j-columns per chunk
    int jb = jc * 32;
    // S = Q(16x64) @ K_j^T  -> relu/scale/exp -> stage P bf16 [16][32]
#pragma unroll
    for (int jt = 0; jt < 2; ++jt) {
      v16bf kb0 = ldB(Ks, jb + jt * 16, 0, 64, lane);
      v16bf kb1 = ldB(Ks, jb + jt * 16, 32, 64, lane);
      v8f s = {};
      s = wmma_bf16(a0, kb0, s);
      s = wmma_bf16(a1, kb1, s);
#pragma unroll
      for (int r = 0; r < 8; ++r) {
        float f = s[r] * SCALE_;
        f = f > 0.f ? f : 0.f;               // relu before softmax
        float p = __expf(f);                 // relu'd scores are small: no max-sub needed
        int row = r + ((lane >> 4) << 3);
        int col = jt * 16 + (lane & 15);
        pw[row * 32 + col] = f2bf(p);        // D-layout -> row-major via LDS
      }
    }
    // re-read P in A layout (same-wave DS ops are in-order)
    v16bf pa = ldA(pw, 0, 0, 32, lane);
    lacc = wmma_bf16(pa, ones, lacc);        // row sums, replicated per lane
#pragma unroll
    for (int ht = 0; ht < 4; ++ht) {
      v16bf vb = ldB(Vts, ht * 16, jb, 512, lane);  // V^T rows contiguous in j
      o[ht] = wmma_bf16(pa, vb, o[ht]);
    }
  }

  // O /= rowsum, optional relu, store fp32 in caller-chosen layout
#pragma unroll
  for (int ht = 0; ht < 4; ++ht) {
#pragma unroll
    for (int r = 0; r < 8; ++r) {
      float val = o[ht][r] / lacc[r];
      if (do_relu) val = val > 0.f ? val : 0.f;
      int row = iBase + r + ((lane >> 4) << 3);
      int col = ht * 16 + (lane & 15);
      out[(size_t)b * soB + (size_t)t * soT + (size_t)row * soN + col] = val;
    }
  }
}

// ---------------------------------------------------------------------------
extern "C" void kernel_launch(void* const* d_in, const int* in_sizes, int n_in,
                              void* d_out, int out_size, void* d_ws, size_t ws_size,
                              hipStream_t stream) {
  const float* x   = (const float*)d_in[0];
  const float* Wq1 = (const float*)d_in[1];  const float* bq1 = (const float*)d_in[2];
  const float* Wk1 = (const float*)d_in[3];  const float* bk1 = (const float*)d_in[4];
  const float* Wv1 = (const float*)d_in[5];  const float* bv1 = (const float*)d_in[6];
  const float* Wq2 = (const float*)d_in[7];  const float* bq2 = (const float*)d_in[8];
  const float* Wk2 = (const float*)d_in[9];  const float* bk2 = (const float*)d_in[10];
  const float* Wv2 = (const float*)d_in[11]; const float* bv2 = (const float*)d_in[12];
  float* outp = (float*)d_out;

  // workspace: Q,K bf16 [B*T][N][H]; V^T bf16 [B*T][H][N]; h1 f32 [B,T,N,H]
  const size_t qElems = (size_t)B_ * T_ * N_ * H_;   // 6.29M
  unsigned short* Qb  = (unsigned short*)d_ws;
  unsigned short* Kb  = Qb + qElems;
  unsigned short* Vtb = Kb + qElems;
  float* h1 = (float*)(Vtb + qElems);                // ~63MB total

  const int projBlocks = (B_ * T_ * N_ / 16) / 4;    // 1536
  const int attnBlocks = B_ * T_ * 4;                // 768
  const size_t attnLds = (size_t)(512 * 64 + 64 * 512 + 8 * 16 * 32) * 2; // 136KB

  // ---- layer 1 ----
  proj_kernel<<<projBlocks, 128, 0, stream>>>(
      x, Wq1, bq1, Wk1, bk1, Wv1, bv1, Qb, Kb, Vtb,
      /*sB=*/N_ * T_ * H_, /*sN=*/T_ * H_, /*sT=*/H_);          // x: [B,N,T,D]
  attn_kernel<<<attnBlocks, 256, attnLds, stream>>>(
      Qb, Kb, Vtb, h1,
      /*soB=*/T_ * N_ * H_, /*soT=*/N_ * H_, /*soN=*/H_, /*relu=*/0); // h1: [B,T,N,H]

  // ---- layer 2 ----
  proj_kernel<<<projBlocks, 128, 0, stream>>>(
      h1, Wq2, bq2, Wk2, bk2, Wv2, bv2, Qb, Kb, Vtb,
      /*sB=*/T_ * N_ * H_, /*sN=*/H_, /*sT=*/N_ * H_);          // h1: [B,T,N,H]
  attn_kernel<<<attnBlocks, 256, attnLds, stream>>>(
      Qb, Kb, Vtb, outp,
      /*soB=*/N_ * T_ * H_, /*soT=*/H_, /*soN=*/T_ * H_, /*relu=*/1); // out: [B,N,T,H]
}
